// Mean_Shift_Cluster_83794811945535
// MI455X (gfx1250) — matrix operand order repaired and checked by reading, not compiled, not run on hardware
//
#include <hip/hip_runtime.h>

typedef __attribute__((ext_vector_type(16))) _Float16 v16h;
typedef __attribute__((ext_vector_type(8)))  _Float16 v8h;
typedef __attribute__((ext_vector_type(4)))  _Float16 v4h;
typedef __attribute__((ext_vector_type(8)))  float    v8f;

#define D_FEAT 256
#define N_PTS  8192
#define QTILE  64          // queries per block (4 waves x 16)
#define KTILE  128         // keys staged per LDS tile
#define NWAVE  4
#define DELTA_LOG2E 8.656170245333781f   // 6.0 * log2(e)

// Fused mean-shift step:  Y[:,j] = sum_i exp(6*x_i.x_j) * x_i / sum_i exp(6*x_i.x_j)
// X, Y are [D_FEAT][N_PTS] row-major fp32 (d-major, matching the JAX layout).
__launch_bounds__(128)
__global__ void ms_iter_kernel(const float* __restrict__ Xsrc, float* __restrict__ Ydst)
{
    __shared__ _Float16 KM[KTILE][D_FEAT];   // key-major staging: KM[key][d]
    __shared__ _Float16 DM[D_FEAT][KTILE];   // d-major staging:   DM[d][key]
    __shared__ _Float16 PB[NWAVE][16 * 32];  // per-wave P (16q x 32k) bounce buffer

    const int tid  = threadIdx.x;
    const int lane = tid & 31;
    const int wv   = tid >> 5;        // 0..3
    const int hf   = lane >> 4;       // lane half (0: lanes 0-15, 1: lanes 16-31)
    const int l16  = lane & 15;

    const int qBase = blockIdx.x * QTILE;     // 128 blocks cover N_PTS
    const int qRow  = qBase + wv * 16;        // this wave's 16 query rows

    // O accumulator: 16 q x 256 d fp32, as 16 C-fragments (v8f each)
    v8f O[16];
#pragma unroll
    for (int i = 0; i < 16; ++i) O[i] = (v8f){};
    float rs[8];
#pragma unroll
    for (int r = 0; r < 8; ++r) rs[r] = 0.0f;

    v16h aq[8];  // Q A-fragments: 16 q x 256 d f16 resident in VGPRs

    const int nkt = N_PTS / KTILE;            // 64 key tiles
    const int kt0 = (int)blockIdx.x >> 1;     // first tile contains our own q rows

    for (int t = 0; t < nkt; ++t) {
        const int kt     = (kt0 + t) & (nkt - 1);
        const int ktBase = kt * KTILE;

        __syncthreads();  // previous tile fully consumed before overwrite
        // ---- cooperative staging: 128 keys x 256 d, f32->f16, dual layout ----
        // each wave stages 64 d-rows; lane covers 4 consecutive keys (coalesced b128 reads)
        for (int i = 0; i < 64; ++i) {
            const int d = wv * 64 + i;
            const float4 f = *reinterpret_cast<const float4*>(
                Xsrc + (size_t)d * N_PTS + ktBase + lane * 4);
            const _Float16 h0 = (_Float16)f.x, h1 = (_Float16)f.y;
            const _Float16 h2 = (_Float16)f.z, h3 = (_Float16)f.w;
            v4h dv = {h0, h1, h2, h3};
            *(v4h*)&DM[d][lane * 4] = dv;          // contiguous b64 store
            KM[lane * 4 + 0][d] = h0;              // scattered b16 stores
            KM[lane * 4 + 1][d] = h1;
            KM[lane * 4 + 2][d] = h2;
            KM[lane * 4 + 3][d] = h3;
        }
        __syncthreads();

        if (t == 0) {
            // Q A-fragments from the tile that contains our q rows.
            // A operand, contiguous k order used here (same as B): lane hf=0: K = c*32 + {0..15};
            //                          lane hf=1: K = c*32 + {16..31}.
            const int qr = ((blockIdx.x & 1) * 64) + wv * 16 + l16;
#pragma unroll
            for (int c = 0; c < 8; ++c) {
                aq[c] = *(const v16h*)&KM[qr][c * 32 + hf * 16];
            }
        }

        // ---- process tile: 4 chunks of 32 keys ----
        for (int kb = 0; kb < KTILE / 32; ++kb) {
            const int key0 = kb * 32;
            v8f s0 = (v8f){}, s1 = (v8f){};
            // S = Q . K^T over 256-d in 8 K=32 chunks.
            // B operand, same contiguous k order as A: lane n = l16; K = hf*16 + 0..15 contiguous in d.
#pragma unroll
            for (int c = 0; c < 8; ++c) {
                const v16h b0 = *(const v16h*)&KM[key0 + l16][c * 32 + hf * 16];
                s0 = __builtin_amdgcn_wmma_f32_16x16x32_f16(
                        false, aq[c], false, b0, (short)0, s0, false, false);
            }
#pragma unroll
            for (int c = 0; c < 8; ++c) {
                const v16h b1 = *(const v16h*)&KM[key0 + 16 + l16][c * 32 + hf * 16];
                s1 = __builtin_amdgcn_wmma_f32_16x16x32_f16(
                        false, aq[c], false, b1, (short)0, s1, false, false);
            }
            // P = exp(6*S); accumulate row sums (per-lane partials over key column l16)
            v8f p0, p1;
#pragma unroll
            for (int r = 0; r < 8; ++r) {
                p0[r] = __builtin_amdgcn_exp2f(s0[r] * DELTA_LOG2E);
                p1[r] = __builtin_amdgcn_exp2f(s1[r] * DELTA_LOG2E);
                rs[r] += p0[r] + p1[r];
            }
            // Bounce P (C layout) through wave-private LDS into A layout (16q x 32k f16)
            _Float16* pb = &PB[wv][0];
#pragma unroll
            for (int r = 0; r < 8; ++r) {
                pb[(r + 8 * hf) * 32 + l16]      = (_Float16)p0[r];
                pb[(r + 8 * hf) * 32 + 16 + l16] = (_Float16)p1[r];
            }
            // DS ops are in-order within a wave; block compiler reordering + drain DS.
            asm volatile("s_wait_dscnt 0" ::: "memory");
            const v16h pa = *(const v16h*)&pb[l16 * 32 + hf * 16];
            // O += P . V  (V rows = keys, 16-d column blocks; B from d-major tile)
#pragma unroll
            for (int c2 = 0; c2 < 16; ++c2) {
                const v16h bv = *(const v16h*)&DM[c2 * 16 + l16][key0 + hf * 16];
                O[c2] = __builtin_amdgcn_wmma_f32_16x16x32_f16(
                        false, pa, false, bv, (short)0, O[c2], false, false);
            }
        }
    }

    // ---- reduce column sums across each 16-lane half (matches C-frag row mapping) ----
#pragma unroll
    for (int m = 0; m < 4; ++m) {
        const int mask = 1 << m;
#pragma unroll
        for (int r = 0; r < 8; ++r) rs[r] += __shfl_xor(rs[r], mask, 32);
    }
    float rinv[8];
#pragma unroll
    for (int r = 0; r < 8; ++r) rinv[r] = 1.0f / rs[r];

    // ---- normalize and store: C-frag lane l: d = c2*16 + l16, q = qRow + 8*hf + r ----
#pragma unroll
    for (int c2 = 0; c2 < 16; ++c2) {
        const int d = c2 * 16 + l16;
#pragma unroll
        for (int r = 0; r < 8; ++r) {
            const int q = qRow + 8 * hf + r;
            Ydst[(size_t)d * N_PTS + q] = O[c2][r] * rinv[r];
        }
    }
}

extern "C" void kernel_launch(void* const* d_in, const int* in_sizes, int n_in,
                              void* d_out, int out_size, void* d_ws, size_t ws_size,
                              hipStream_t stream) {
    (void)in_sizes; (void)n_in; (void)d_ws; (void)ws_size; (void)out_size;
    const float* X = (const float*)d_in[0];
    float* out = (float*)d_out;
    const size_t len = (size_t)D_FEAT * N_PTS;
    float* Y0 = out;
    float* Y1 = out + len;
    float* Y2 = out + 2 * len;

    dim3 grid(N_PTS / QTILE);   // 128 blocks
    dim3 block(128);            // 4 waves (wave32)
    hipLaunchKernelGGL(ms_iter_kernel, grid, block, 0, stream, X,  Y0);
    hipLaunchKernelGGL(ms_iter_kernel, grid, block, 0, stream, Y0, Y1);
    hipLaunchKernelGGL(ms_iter_kernel, grid, block, 0, stream, Y1, Y2);
}
